// RNN_0_28389733826915
// MI455X (gfx1250) — matrix operand order repaired and checked by reading, not compiled, not run on hardware
//
#include <hip/hip_runtime.h>
#include <type_traits>

typedef __attribute__((ext_vector_type(16))) _Float16 v16h;
typedef __attribute__((ext_vector_type(8)))  _Float16 v8h;
typedef __attribute__((ext_vector_type(8)))  float    v8f;
typedef __attribute__((ext_vector_type(4)))  float    v4f;

namespace {
constexpr int NB = 8192;   // batch
constexpr int NT = 128;    // timesteps
constexpr int NI = 28;     // input features (padded to K=32)
constexpr int NH = 64;     // hidden
constexpr int NO = 10;     // output
constexpr int ROWS = 16;   // batch rows per wave (WMMA M)
}

__device__ __forceinline__ float hw_tanhf(float x) {
#if __has_builtin(__builtin_amdgcn_tanhf)
  return __builtin_amdgcn_tanhf(x);                 // v_tanh_f32 (CDNA5 TRANS op)
#elif __has_builtin(__builtin_amdgcn_tanh_f32)
  return __builtin_amdgcn_tanh_f32(x);
#else
  // Branchless: tanh(x) = sign(x)*(1-t)*rcp(1+t), t = exp2(-2*log2(e)*|x|).
  // Large |x| -> t underflows to 0 -> result +/-1. No exec divergence.
  const float ax = __builtin_fabsf(x);
  const float t  = __builtin_amdgcn_exp2f(-2.8853900817779268f * ax);
  const float r  = (1.0f - t) * __builtin_amdgcn_rcpf(1.0f + t);
  return __builtin_copysignf(r, x);
#endif
}

__global__ __launch_bounds__(32) void rnn_elman_fused(
    const float* __restrict__ x,
    const float* __restrict__ W_ih,
    const float* __restrict__ W_hh,
    const float* __restrict__ b_ih,
    const float* __restrict__ b_hh,
    const float* __restrict__ W_out,
    const float* __restrict__ b_out,
    float* __restrict__ out)
{
  __shared__ __align__(16) _Float16 hbuf[ROWS * NH];   // h_t (f16), row-major [16][64]
  __shared__ __align__(16) float    hlast[ROWS * NH];  // final-step h (f32)

  const int lane = threadIdx.x & 31;
  const int half = lane >> 4;    // 0: lanes 0-15, 1: lanes 16-31
  const int l16  = lane & 15;
  const long row0 = (long)blockIdx.x * ROWS;

  // ---- B fragments for W_ih^T (K=32 padded from 28; four N-tiles of 16) ----
  // B layout (16x16x32 f16): lanes 0-15: N=lane,    local K = {0..7, 16..23} (2/VGPR)
  //                          lanes 16-31: N=lane-16, local K = {8..15, 24..31}
  v16h Bih[4];
#pragma unroll
  for (int n = 0; n < 4; ++n) {
    const float* wr = W_ih + (n * 16 + l16) * NI;
    const int kb = half * 8;
#pragma unroll
    for (int j = 0; j < 16; ++j) {
      const int k = kb + j + (j >= 8 ? 8 : 0);
      Bih[n][j] = (k < NI) ? (_Float16)wr[k] : (_Float16)0.f;
    }
  }

  // ---- B fragments for W_hh^T (K=64 -> two 32-chunks; four N-tiles) ----
  v16h Bhh[4][2];
#pragma unroll
  for (int n = 0; n < 4; ++n) {
#pragma unroll
    for (int kh = 0; kh < 2; ++kh) {
      const float* wr = W_hh + (n * 16 + l16) * NH + kh * 32 + half * 8;
#pragma unroll
      for (int j = 0; j < 16; ++j) Bhh[n][kh][j] = (_Float16)wr[j + (j >= 8 ? 8 : 0)];
    }
  }

  // per-lane fused bias: column = n*16 + lane%16, same for all 8 rows of C/D
  float bias[4];
#pragma unroll
  for (int n = 0; n < 4; ++n) {
    const int c = n * 16 + l16;
    bias[n] = b_ih[c] + b_hh[c];
  }

  // h_0 = 0
  for (int i = lane; i < ROWS * NH; i += 32) hbuf[i] = (_Float16)0.f;
  __builtin_amdgcn_wave_barrier();

  const float* xrow = x + (row0 + l16) * (long)NT * NI;

  auto step = [&](int t, auto last_c) {
    constexpr bool LAST = decltype(last_c)::value;

    const float* xp = xrow + t * NI;
    __builtin_prefetch(xp + NI, 0, 1);   // global_prefetch_b8 (speculative; OOB dropped)

    // ---- A fragments of h_{t-1} from LDS (two K-chunks of 32) ----
    // A layout (16x32 f16): lanes 0-15: M=lane,    K = {0..7, 16..23}
    //                       lanes 16-31: M=lane-16, K = {8..15, 24..31}
    v16h Ah[2];
#pragma unroll
    for (int ka = 0; ka < 2; ++ka) {
      const v8h* hp = (const v8h*)(hbuf + l16 * NH + ka * 32 + half * 8);
      const v8h c0 = hp[0];   // local K base + 0..7   (ds_load_b128)
      const v8h c1 = hp[2];   // local K base + 16..23 (ds_load_b128)
#pragma unroll
      for (int j = 0; j < 8; ++j) { Ah[ka][j] = c0[j]; Ah[ka][8 + j] = c1[j]; }
    }

    // ---- A fragment of x_t, branchless (K padded 28 -> 32) ----
    // 4 unconditional float4 loads, all in-bounds for both lane halves;
    // the K=28..31 pad is a per-lane cndmask select, not a branch.
    const v4f* xv = (const v4f*)xp;               // 16B-aligned: 28 & row stride % 4 == 0
    const v4f a0 = xv[half * 2 + 0];              // floats  0..3  |  8..11
    const v4f a1 = xv[half * 2 + 1];              // floats  4..7  | 12..15
    const v4f e0 = xv[4 + half * 2];              // floats 16..19 | 24..27
    const v4f e1 = xv[5 + half];                  // floats 20..23 | 24..27 (dummy)
    v16h Ax;
#pragma unroll
    for (int j = 0; j < 4; ++j) {
      Ax[j]      = (_Float16)a0[j];
      Ax[4 + j]  = (_Float16)a1[j];
      Ax[8 + j]  = (_Float16)e0[j];
      Ax[12 + j] = half ? (_Float16)0.f : (_Float16)e1[j];
    }

    __builtin_amdgcn_wave_barrier();  // reads of h_{t-1} precede writes of h_t

    // ---- h_t = tanh(x_t W_ih^T + b + h_{t-1} W_hh^T), four 16-column tiles ----
#pragma unroll
    for (int n = 0; n < 4; ++n) {
      v8f c = {};
      c = __builtin_amdgcn_wmma_f32_16x16x32_f16(false, Ax,    false, Bih[n],
                                                 (short)0, c, false, false);
      c = __builtin_amdgcn_wmma_f32_16x16x32_f16(false, Ah[0], false, Bhh[n][0],
                                                 (short)0, c, false, false);
      c = __builtin_amdgcn_wmma_f32_16x16x32_f16(false, Ah[1], false, Bhh[n][1],
                                                 (short)0, c, false, false);
      const int col = n * 16 + l16;
#pragma unroll
      for (int r = 0; r < 8; ++r) {              // C/D layout: M = half*8 + r
        const float v = hw_tanhf(c[r] + bias[n]);
        const int m = half * 8 + r;
        hbuf[m * NH + col] = (_Float16)v;
        if (LAST) hlast[m * NH + col] = v;
      }
    }
    __builtin_amdgcn_wave_barrier();
  };

  for (int t = 0; t < NT - 1; ++t)
    step(t, std::integral_constant<bool, false>{});
  step(NT - 1, std::integral_constant<bool, true>{});

  __builtin_amdgcn_wave_barrier();
  // ---- out = h_T @ W_out^T + b_out : 16x10 results, 5 per lane ----
  for (int p = lane; p < ROWS * NO; p += 32) {
    const int r = p / NO, o = p % NO;
    const float* hr = hlast + r * NH;
    const float* wo = W_out + o * NH;
    float acc = b_out[o];
#pragma unroll
    for (int k = 0; k < NH; ++k) acc = fmaf(hr[k], wo[k], acc);
    out[(row0 + r) * NO + o] = acc;
  }
}

extern "C" void kernel_launch(void* const* d_in, const int* /*in_sizes*/, int /*n_in*/,
                              void* d_out, int /*out_size*/, void* /*d_ws*/, size_t /*ws_size*/,
                              hipStream_t stream) {
  const float* x     = (const float*)d_in[0];
  const float* W_ih  = (const float*)d_in[1];
  const float* W_hh  = (const float*)d_in[2];
  const float* b_ih  = (const float*)d_in[3];
  const float* b_hh  = (const float*)d_in[4];
  const float* W_out = (const float*)d_in[5];
  const float* b_out = (const float*)d_in[6];
  float* out = (float*)d_out;

  dim3 grid(NB / ROWS), block(32);
  rnn_elman_fused<<<grid, block, 0, stream>>>(x, W_ih, W_hh, b_ih, b_hh,
                                              W_out, b_out, out);
}
